// SelfAttention2d_8392366097136
// MI455X (gfx1250) — matrix operand order repaired and checked by reading, not compiled, not run on hardware
//
#include <hip/hip_runtime.h>

typedef __attribute__((ext_vector_type(16))) _Float16 v16h;
typedef __attribute__((ext_vector_type(8)))  _Float16 v8h;
typedef __attribute__((ext_vector_type(8)))  float    v8f;

#define B_  4
#define C_  256
#define T_  2304
#define NH_ 8
#define HD_ 32
#define G_  8
#define CPG_ 32

// ---------------- WMMA fragment helpers (wave32, 16x16x32 f16 -> f32) ----------------

__device__ __forceinline__ v8f wmma32(v16h a, v16h b, v8f c) {
  return __builtin_amdgcn_wmma_f32_16x16x32_f16(
      /*neg_a=*/false, a, /*neg_b=*/false, b,
      /*c_mod=*/(short)0, c, /*reuse_a=*/false, /*reuse_b=*/false);
}

// A fragment: 16(M) x 32(K) f16 from row-major [row][k] storage.
__device__ __forceinline__ v16h load_a16(const _Float16* __restrict__ base,
                                         int row, int stride, int kc, int lane) {
  const _Float16* p = base + (size_t)row * stride + kc + ((lane & 16) ? 8 : 0);
  v8h lo = *reinterpret_cast<const v8h*>(p);
  v8h hi = *reinterpret_cast<const v8h*>(p + 16);
  v16h r;
#pragma unroll
  for (int i = 0; i < 8; ++i) { r[i] = lo[i]; r[i + 8] = hi[i]; }
  return r;
}

// B fragment: 32(K) x 16(N) f16; storage [col][k], each column's K contiguous.
__device__ __forceinline__ v16h load_b16(const _Float16* __restrict__ base,
                                         int col, int stride, int kc, int lane) {
  const _Float16* p = base + (size_t)col * stride + kc + ((lane & 16) ? 8 : 0);
  v8h lo = *reinterpret_cast<const v8h*>(p);
  v8h hi = *reinterpret_cast<const v8h*>(p + 16);
  v16h r;
#pragma unroll
  for (int i = 0; i < 8; ++i) { r[i] = lo[i]; r[i + 8] = hi[i]; }
  return r;
}

__device__ __forceinline__ float halfmax(float v) {
#pragma unroll
  for (int m = 1; m <= 8; m <<= 1) v = fmaxf(v, __shfl_xor(v, m, 32));
  return v;
}
__device__ __forceinline__ float halfsum(float v) {
#pragma unroll
  for (int m = 1; m <= 8; m <<= 1) v += __shfl_xor(v, m, 32);
  return v;
}

// CDNA5 async global->LDS copy, 16B per lane, tracked by ASYNCcnt.
__device__ __forceinline__ void async_copy16(unsigned lds_off, const void* gptr) {
  unsigned long long ga = (unsigned long long)(uintptr_t)gptr;
  asm volatile("global_load_async_to_lds_b128 %0, %1, off"
               :: "v"(lds_off), "v"(ga) : "memory");
}

// ---------------- 0) weight f32 -> f16 convert ----------------
__global__ void __launch_bounds__(256)
convert_w_kernel(const float* __restrict__ wq, const float* __restrict__ wp,
                 _Float16* __restrict__ wq16, _Float16* __restrict__ wp16) {
  int i = blockIdx.x * 256 + threadIdx.x;
  if (i < 3 * C_ * C_) wq16[i] = (_Float16)wq[i];
  if (i < C_ * C_)     wp16[i] = (_Float16)wp[i];
}

// ---------------- 1) GroupNorm -> f16 [b][t][c] ----------------
__global__ void __launch_bounds__(256)
gn_kernel(const float* __restrict__ x, const float* __restrict__ w,
          const float* __restrict__ bsc, _Float16* __restrict__ n16) {
  const float EPS = 1e-5f;
  int bg = blockIdx.x;
  int b = bg >> 3, g = bg & 7;
  int tid = threadIdx.x;
  const int NELT = CPG_ * T_;
  const float* xp = x + ((size_t)b * C_ + g * CPG_) * T_;

  float s = 0.f, sq = 0.f;
  for (int i = tid; i < NELT; i += 256) { float v = xp[i]; s += v; sq += v * v; }

  __shared__ float ssum[256], ssq[256];
  ssum[tid] = s; ssq[tid] = sq;
  __syncthreads();
  for (int st = 128; st > 0; st >>= 1) {
    if (tid < st) { ssum[tid] += ssum[tid + st]; ssq[tid] += ssq[tid + st]; }
    __syncthreads();
  }
  float mean = ssum[0] / (float)NELT;
  float var  = ssq[0] / (float)NELT - mean * mean;
  float rstd = rsqrtf(var + EPS);

  for (int i = tid; i < NELT; i += 256) {
    int c = g * CPG_ + (i & 31);
    int t = i >> 5;
    float v  = x[((size_t)b * C_ + c) * T_ + t];
    float nv = (v - mean) * rstd * w[c] + bsc[c];
    n16[((size_t)b * T_ + t) * C_ + c] = (_Float16)nv;
  }
}

// ---------------- 2) QKV GEMM: waves in a block share the B (activation) tile ----------------
__global__ void __launch_bounds__(256)
qkv_kernel(const _Float16* __restrict__ W, const float* __restrict__ bias,
           const _Float16* __restrict__ n16, _Float16* __restrict__ q16,
           _Float16* __restrict__ k16, _Float16* __restrict__ v16) {
  int lane = threadIdx.x & 31;
  int wib = threadIdx.x >> 5;
  int bx = blockIdx.x;
  int tT = bx % 36; bx /= 36;          // shared by all 8 waves -> B tile in WGP$
  int oBlk = bx % 6; int b = bx / 6;
  int oT = oBlk * 8 + wib;             // 48 row-tiles of 16
  int ln = lane & 15, hl = (lane >> 4) & 1;
  const _Float16* Bb = n16 + (size_t)b * T_ * C_;

  v8f acc[4] = {{}, {}, {}, {}};
  for (int kc = 0; kc < C_; kc += 32) {
    v16h a = load_a16(W, oT * 16 + ln, C_, kc, lane);
#pragma unroll
    for (int j = 0; j < 4; ++j) {
      v16h bf = load_b16(Bb, tT * 64 + j * 16 + ln, C_, kc, lane);
      acc[j] = wmma32(a, bf, acc[j]);
    }
  }
#pragma unroll
  for (int j = 0; j < 4; ++j) {
#pragma unroll
    for (int r = 0; r < 8; ++r) {
      int o = oT * 16 + r + hl * 8;
      int t = tT * 64 + j * 16 + ln;
      float v = acc[j][r] + bias[o];
      int sec = o >> 8;
      int ch = o & 255;
      int h = ch >> 5, d = ch & 31;
      _Float16 hv = (_Float16)v;
      if (sec == 0)      q16[(((size_t)b * NH_ + h) * T_ + t) * HD_ + d] = hv;
      else if (sec == 1) k16[(((size_t)b * NH_ + h) * T_ + t) * HD_ + d] = hv;
      else               v16[(((size_t)b * NH_ + h) * HD_ + d) * T_ + t] = hv;
    }
  }
}

// ---------------- 3) Flash attention, block-cooperative with async K/V staging ----------------
// Block = 8 waves, one (b,h), 8 consecutive 16-query tiles. Keys streamed in
// 64-wide chunks staged via global_load_async_to_lds_b128, double-buffered.
__global__ void __launch_bounds__(256)
attn_kernel(const _Float16* __restrict__ q16, const _Float16* __restrict__ k16,
            const _Float16* __restrict__ v16, _Float16* __restrict__ ao16) {
  const float scale = 0.17677669529663687f;   // 1/sqrt(32)
  __shared__ __align__(16) _Float16 Kbuf[2][64 * 32];  // [s][d], d contiguous
  __shared__ __align__(16) _Float16 Vbuf[2][32 * 64];  // [d][s], s contiguous
  __shared__ __align__(16) _Float16 Pt[8][16 * 64];

  int tid = threadIdx.x;
  int lane = tid & 31;
  int wib = tid >> 5;
  int ln = lane & 15, hl = (lane >> 4) & 1;
  int bh = blockIdx.x / 18;                    // 32 (b*8+h)
  int tT = (blockIdx.x % 18) * 8 + wib;        // 144 query tiles

  const _Float16* Q = q16 + (size_t)bh * T_ * HD_;
  const _Float16* K = k16 + (size_t)bh * T_ * HD_;
  const _Float16* V = v16 + (size_t)bh * HD_ * T_;
  _Float16* P = &Pt[wib][0];

  // staging: 256 lanes x 16B = 4KB per region per chunk
  const int vrow = tid >> 3, vcg = (tid & 7) * 8;
  auto stage = [&](int buf, int kc) {
    unsigned kdst = (unsigned)(uintptr_t)(&Kbuf[buf][0]) + tid * 16;
    async_copy16(kdst, (const char*)(K + (size_t)kc * HD_) + tid * 16);
    unsigned vdst = (unsigned)(uintptr_t)(&Vbuf[buf][0]) + (vrow * 64 + vcg) * 2;
    async_copy16(vdst, V + (size_t)vrow * T_ + kc + vcg);
  };

  v16h qf = load_a16(Q, tT * 16 + ln, HD_, 0, lane);
  v8f accA = {}; v8f accB = {};
  float m[8], l[8];
#pragma unroll
  for (int r = 0; r < 8; ++r) { m[r] = -1e30f; l[r] = 0.f; }

  stage(0, 0);
  for (int it = 0; it < T_ / 64; ++it) {
    int buf = it & 1;
    asm volatile("s_wait_asynccnt 0" ::: "memory");
    __syncthreads();
    if (it + 1 < T_ / 64) stage(buf ^ 1, (it + 1) * 64);

    const _Float16* Kl = &Kbuf[buf][0];
    const _Float16* Vl = &Vbuf[buf][0];

    // scores: 4 x 16-key subtiles
    v8f s[4];
#pragma unroll
    for (int j = 0; j < 4; ++j) {
      v16h kb = load_b16(Kl, j * 16 + ln, HD_, 0, lane);
      v8f z = {};
      s[j] = wmma32(qf, kb, z);
    }

    // online softmax over 64 columns
#pragma unroll
    for (int r = 0; r < 8; ++r) {
      float a0 = s[0][r] * scale, a1 = s[1][r] * scale;
      float a2 = s[2][r] * scale, a3 = s[3][r] * scale;
      float mx = halfmax(fmaxf(fmaxf(a0, a1), fmaxf(a2, a3)));
      float mn = fmaxf(m[r], mx);
      float alpha = __expf(m[r] - mn);
      m[r] = mn;
      float p0 = __expf(a0 - mn), p1 = __expf(a1 - mn);
      float p2 = __expf(a2 - mn), p3 = __expf(a3 - mn);
      l[r] = l[r] * alpha + halfsum(p0 + p1 + p2 + p3);
      accA[r] *= alpha; accB[r] *= alpha;
      int row = r + hl * 8;
      P[row * 64 + ln]      = (_Float16)p0;
      P[row * 64 + 16 + ln] = (_Float16)p1;
      P[row * 64 + 32 + ln] = (_Float16)p2;
      P[row * 64 + 48 + ln] = (_Float16)p3;
    }
    asm volatile("s_wait_dscnt 0" ::: "memory");

    // PV: out(16x32) += P(16x64) x V(64x32)
    v16h pf0 = load_a16(P, ln, 64, 0, lane);
    v16h pf1 = load_a16(P, ln, 64, 32, lane);
    v16h v00 = load_b16(Vl, ln, 64, 0, lane);
    v16h v01 = load_b16(Vl, ln, 64, 32, lane);
    v16h v10 = load_b16(Vl, 16 + ln, 64, 0, lane);
    v16h v11 = load_b16(Vl, 16 + ln, 64, 32, lane);
    accA = wmma32(pf0, v00, accA);
    accA = wmma32(pf1, v01, accA);
    accB = wmma32(pf0, v10, accB);
    accB = wmma32(pf1, v11, accB);
  }

  int b = bh >> 3, h = bh & 7;
#pragma unroll
  for (int r = 0; r < 8; ++r) {
    int t = tT * 16 + r + hl * 8;
    float inv = 1.0f / l[r];
    size_t o = ((size_t)b * T_ + t) * C_ + h * 32;
    ao16[o + ln]      = (_Float16)(accA[r] * inv);
    ao16[o + 16 + ln] = (_Float16)(accB[r] * inv);
  }
}

// ---------------- 4) proj GEMM + bias + residual (fp32 out) ----------------
__global__ void __launch_bounds__(256)
proj_kernel(const _Float16* __restrict__ W, const float* __restrict__ bias,
            const _Float16* __restrict__ ao16, const float* __restrict__ x,
            float* __restrict__ out) {
  int lane = threadIdx.x & 31;
  int wib = threadIdx.x >> 5;
  int bx = blockIdx.x;
  int tT = bx % 36; bx /= 36;
  int oBlk = bx % 2; int b = bx / 2;
  int oT = oBlk * 8 + wib;             // 16 row-tiles of 16
  int ln = lane & 15, hl = (lane >> 4) & 1;
  const _Float16* Bb = ao16 + (size_t)b * T_ * C_;

  v8f acc[4] = {{}, {}, {}, {}};
  for (int kc = 0; kc < C_; kc += 32) {
    v16h a = load_a16(W, oT * 16 + ln, C_, kc, lane);
#pragma unroll
    for (int j = 0; j < 4; ++j) {
      v16h bf = load_b16(Bb, tT * 64 + j * 16 + ln, C_, kc, lane);
      acc[j] = wmma32(a, bf, acc[j]);
    }
  }
#pragma unroll
  for (int j = 0; j < 4; ++j) {
#pragma unroll
    for (int r = 0; r < 8; ++r) {
      int o = oT * 16 + r + hl * 8;
      int t = tT * 64 + j * 16 + ln;
      size_t idx = ((size_t)b * C_ + o) * T_ + t;
      out[idx] = x[idx] + bias[o] + acc[j][r];
    }
  }
}

// ---------------- host launcher ----------------
extern "C" void kernel_launch(void* const* d_in, const int* in_sizes, int n_in,
                              void* d_out, int out_size, void* d_ws, size_t ws_size,
                              hipStream_t stream) {
  (void)in_sizes; (void)n_in; (void)out_size; (void)ws_size;
  const float* x     = (const float*)d_in[0];
  const float* gnw   = (const float*)d_in[1];
  const float* gnb   = (const float*)d_in[2];
  const float* wqkv  = (const float*)d_in[3];
  const float* bqkv  = (const float*)d_in[4];
  const float* wproj = (const float*)d_in[5];
  const float* bproj = (const float*)d_in[6];
  float* out = (float*)d_out;

  char* ws = (char*)d_ws;
  const size_t SZ_WQ = (size_t)3 * C_ * C_ * 2;
  const size_t SZ_WP = (size_t)C_ * C_ * 2;
  const size_t SZ_BT = (size_t)B_ * T_ * C_ * 2;
  _Float16* wq16 = (_Float16*)(ws);
  _Float16* wp16 = (_Float16*)(ws + SZ_WQ);
  _Float16* n16  = (_Float16*)(ws + SZ_WQ + SZ_WP);
  _Float16* q16  = (_Float16*)(ws + SZ_WQ + SZ_WP + SZ_BT);
  _Float16* k16  = (_Float16*)(ws + SZ_WQ + SZ_WP + 2 * SZ_BT);
  _Float16* v16  = (_Float16*)(ws + SZ_WQ + SZ_WP + 3 * SZ_BT);
  _Float16* ao16 = (_Float16*)(ws + SZ_WQ + SZ_WP + 4 * SZ_BT);

  dim3 blk(256);
  convert_w_kernel<<<dim3((3 * C_ * C_ + 255) / 256), blk, 0, stream>>>(wqkv, wproj, wq16, wp16);
  gn_kernel<<<dim3(B_ * G_), blk, 0, stream>>>(x, gnw, gnb, n16);
  qkv_kernel<<<dim3(864), blk, 0, stream>>>(wq16, bqkv, n16, q16, k16, v16);
  attn_kernel<<<dim3(576), blk, 0, stream>>>(q16, k16, v16, ao16);
  proj_kernel<<<dim3(288), blk, 0, stream>>>(wp16, bproj, ao16, x, out);
}
